// SimpleMamba_8641474200044
// MI455X (gfx1250) — matrix operand order repaired and checked by reading, not compiled, not run on hardware
//
#include <hip/hip_runtime.h>
#include <math.h>

// ---------------------------------------------------------------------------
// Types for CDNA5 WMMA (wave32): v_wmma_f32_16x16x32_bf16
// ---------------------------------------------------------------------------
typedef __attribute__((ext_vector_type(16))) __bf16 v16bf;
typedef __attribute__((ext_vector_type(8)))  __bf16 bf16x8;
typedef __attribute__((ext_vector_type(8)))  float  v8f;

union FragU { v16bf v; bf16x8 h[2]; };

static __device__ __forceinline__ unsigned short f2bf(float f) {
    unsigned int u = __float_as_uint(f);
    u += 0x7fffu + ((u >> 16) & 1u);          // round-to-nearest-even
    return (unsigned short)(u >> 16);
}
static __device__ __forceinline__ float bf2f(unsigned short s) {
    return __uint_as_float(((unsigned int)s) << 16);
}

// Problem constants
#define BB   4
#define TT   4096
#define DD   1024
#define NC   16              // chunks along T
#define LC   (TT / NC)       // 256 steps per chunk
#define MM   (BB * TT)       // 16384 GEMM rows

// ---------------------------------------------------------------------------
// Scan pass 1: per (b, chunk, d) compute chunk-local end state with h0 = 0,
// and emit x in bf16 for the gate GEMM.  65536 threads, coalesced along d.
// ---------------------------------------------------------------------------
__global__ __launch_bounds__(256) void scan_pass1(
    const float* __restrict__ x, const float* __restrict__ A,
    const float* __restrict__ Bp, unsigned short* __restrict__ xbf,
    float* __restrict__ hend)
{
    int i = blockIdx.x * 256 + threadIdx.x;      // [0, BB*NC*DD)
    int d = i & (DD - 1);
    int c = (i >> 10) & (NC - 1);
    int b = i >> 14;
    float a  = A[d];
    float bp = Bp[d];
    size_t base = ((size_t)(b * TT + c * LC)) * DD + d;
    const float* xp = x + base;
    unsigned short* xo = xbf + base;
    float h = 0.0f;
    for (int t = 0; t < LC; ++t) {
        float xv = xp[(size_t)t * DD];
        h = fmaf(a, h, bp * xv);
        xo[(size_t)t * DD] = f2bf(xv);
    }
    hend[i] = h;
}

// ---------------------------------------------------------------------------
// Scan pass 2: combine chunk end states into per-chunk carries.
// carry[c] = H_{c-1},  H_c = A^LC * H_{c-1} + E_c.   4096 threads.
// ---------------------------------------------------------------------------
__global__ __launch_bounds__(256) void scan_pass2(
    const float* __restrict__ A, const float* __restrict__ hend,
    float* __restrict__ carry)
{
    int i = blockIdx.x * 256 + threadIdx.x;      // [0, BB*DD)
    int d = i & (DD - 1);
    int b = i >> 10;
    float alc = A[d];
    #pragma unroll
    for (int j = 0; j < 8; ++j) alc *= alc;      // A^256 (LC = 2^8)
    float cr = 0.0f;
    for (int c = 0; c < NC; ++c) {
        int idx = (b * NC + c) * DD + d;
        carry[idx] = cr;
        cr = fmaf(alc, cr, hend[idx]);
    }
}

// ---------------------------------------------------------------------------
// Scan pass 3: replay each chunk with the correct carry, write ssm = Cp*h (bf16)
// ---------------------------------------------------------------------------
__global__ __launch_bounds__(256) void scan_pass3(
    const float* __restrict__ x, const float* __restrict__ A,
    const float* __restrict__ Bp, const float* __restrict__ Cp,
    const float* __restrict__ carry, unsigned short* __restrict__ ssm)
{
    int i = blockIdx.x * 256 + threadIdx.x;      // [0, BB*NC*DD)
    int d = i & (DD - 1);
    int c = (i >> 10) & (NC - 1);
    int b = i >> 14;
    float a  = A[d];
    float bp = Bp[d];
    float cp = Cp[d];
    size_t base = ((size_t)(b * TT + c * LC)) * DD + d;
    const float* xp = x + base;
    unsigned short* so = ssm + base;
    float h = carry[i];
    for (int t = 0; t < LC; ++t) {
        float xv = xp[(size_t)t * DD];
        h = fmaf(a, h, bp * xv);
        so[(size_t)t * DD] = f2bf(cp * h);
    }
}

// ---------------------------------------------------------------------------
// Convert both weight matrices to bf16
// ---------------------------------------------------------------------------
__global__ __launch_bounds__(256) void convert_w(
    const float* __restrict__ gw, const float* __restrict__ ow,
    unsigned short* __restrict__ gwb, unsigned short* __restrict__ owb)
{
    int i = blockIdx.x * 256 + threadIdx.x;      // [0, DD*DD)
    gwb[i] = f2bf(gw[i]);
    owb[i] = f2bf(ow[i]);
}

// ---------------------------------------------------------------------------
// WMMA GEMM: out[m, n] = sum_k A[m, k] * W[n, k] + bias[n], with epilogue.
// Block = 256 threads = 8 waves (4 m-waves x 2 n-waves).
// Wave tile = 32x64 (2 m-subtiles x 4 n-subtiles of 16x16).
// Block tile = 128x128.  K step = 32 (v_wmma_f32_16x16x32_bf16).
//
// Fragment layouts per CDNA5 ISA 7.12.2 (wave32, half = lane>>4, l15 = lane&15):
//   A 16x32 bf16: lane row m=l15; elems j<8 -> K = k + half*8 + j,
//                                 j>=8 -> K = k + 16 + half*8 + (j-8)
//   B 32x16 bf16: lane col n=l15; same K mapping as A
//   C/D 16x16 f32: lane col n=l15; vgpr v -> row m = half*8 + v
// ---------------------------------------------------------------------------
template <bool GATE>
__global__ __launch_bounds__(256) void gemm_bf16(
    const unsigned short* __restrict__ Abuf,   // MM x DD bf16, row-major
    const unsigned short* __restrict__ Wbuf,   // DD x DD bf16, row-major (n, k)
    const float* __restrict__ bias,            // DD
    const unsigned short* __restrict__ ssm,    // MM x DD bf16 (GATE only)
    unsigned short* __restrict__ ybf,          // MM x DD bf16 (GATE only)
    float* __restrict__ outf)                  // MM x DD f32  (!GATE only)
{
    constexpr int K = DD;
    constexpr int N = DD;

    const int lane = threadIdx.x & 31;
    const int wave = threadIdx.x >> 5;         // 0..7
    const int wm   = wave >> 1;                // 0..3
    const int wn   = wave & 1;                 // 0..1
    const int half = lane >> 4;
    const int l15  = lane & 15;

    const int m0 = blockIdx.x * 128 + wm * 32;
    const int n0 = blockIdx.y * 128 + wn * 64;

    const __bf16* Ab = (const __bf16*)Abuf;
    const __bf16* Wb = (const __bf16*)Wbuf;

    const __bf16* arow[2];
    arow[0] = Ab + (size_t)(m0 + l15) * K;
    arow[1] = arow[0] + (size_t)16 * K;
    const __bf16* brow[4];
    #pragma unroll
    for (int j = 0; j < 4; ++j)
        brow[j] = Wb + (size_t)(n0 + j * 16 + l15) * K;

    v8f acc[2][4];
    #pragma unroll
    for (int ti = 0; ti < 2; ++ti)
        #pragma unroll
        for (int tj = 0; tj < 4; ++tj)
            acc[ti][tj] = (v8f){0.f, 0.f, 0.f, 0.f, 0.f, 0.f, 0.f, 0.f};

    #pragma unroll 2
    for (int k = 0; k < K; k += 32) {
        FragU a[2], b[4];
        #pragma unroll
        for (int ti = 0; ti < 2; ++ti) {
            a[ti].h[0] = *(const bf16x8*)(arow[ti] + k + half * 8);
            a[ti].h[1] = *(const bf16x8*)(arow[ti] + k + 16 + half * 8);
        }
        #pragma unroll
        for (int tj = 0; tj < 4; ++tj) {
            b[tj].h[0] = *(const bf16x8*)(brow[tj] + k + half * 8);
            b[tj].h[1] = *(const bf16x8*)(brow[tj] + k + 16 + half * 8);
        }
        #pragma unroll
        for (int ti = 0; ti < 2; ++ti)
            #pragma unroll
            for (int tj = 0; tj < 4; ++tj)
                acc[ti][tj] = __builtin_amdgcn_wmma_f32_16x16x32_bf16(
                    false, a[ti].v, false, b[tj].v,
                    (short)0, acc[ti][tj], false, false);
    }

    // Epilogue
    #pragma unroll
    for (int ti = 0; ti < 2; ++ti) {
        #pragma unroll
        for (int tj = 0; tj < 4; ++tj) {
            const int n = n0 + tj * 16 + l15;
            const float bia = bias[n];
            const int mbase = m0 + ti * 16 + half * 8;
            #pragma unroll
            for (int v = 0; v < 8; ++v) {
                const int m = mbase + v;
                const size_t idx = (size_t)m * N + n;
                float val = acc[ti][tj][v] + bia;
                if (GATE) {
                    float g = 1.0f / (1.0f + __expf(-val));
                    ybf[idx] = f2bf(g * bf2f(ssm[idx]));
                } else {
                    outf[idx] = val;
                }
            }
        }
    }
}

// ---------------------------------------------------------------------------
// Launch
// ---------------------------------------------------------------------------
extern "C" void kernel_launch(void* const* d_in, const int* in_sizes, int n_in,
                              void* d_out, int out_size, void* d_ws, size_t ws_size,
                              hipStream_t stream) {
    const float* x  = (const float*)d_in[0];
    const float* A  = (const float*)d_in[1];
    const float* Bp = (const float*)d_in[2];
    const float* Cp = (const float*)d_in[3];
    const float* gw = (const float*)d_in[4];
    const float* gb = (const float*)d_in[5];
    const float* ow = (const float*)d_in[6];
    const float* ob = (const float*)d_in[7];
    float* out = (float*)d_out;

    const size_t nBTD = (size_t)BB * TT * DD;      // 16,777,216
    char* ws = (char*)d_ws;
    unsigned short* xbf  = (unsigned short*)ws;  ws += nBTD * 2;
    unsigned short* ssm  = (unsigned short*)ws;  ws += nBTD * 2;
    unsigned short* ybf  = (unsigned short*)ws;  ws += nBTD * 2;
    unsigned short* gwb  = (unsigned short*)ws;  ws += (size_t)DD * DD * 2;
    unsigned short* owb  = (unsigned short*)ws;  ws += (size_t)DD * DD * 2;
    float* hend  = (float*)ws;                   ws += (size_t)BB * NC * DD * 4;
    float* carry = (float*)ws;

    // SSM scan (chunked, 3 passes) + bf16 conversion of x
    scan_pass1<<<(BB * NC * DD) / 256, 256, 0, stream>>>(x, A, Bp, xbf, hend);
    scan_pass2<<<(BB * DD) / 256, 256, 0, stream>>>(A, hend, carry);
    scan_pass3<<<(BB * NC * DD) / 256, 256, 0, stream>>>(x, A, Bp, Cp, carry, ssm);

    // Weights -> bf16
    convert_w<<<(DD * DD) / 256, 256, 0, stream>>>(gw, ow, gwb, owb);

    // GEMM1: gate = sigmoid(x @ gate_w^T + gate_b); y = gate * ssm  (bf16 out)
    dim3 grid(MM / 128, DD / 128);
    gemm_bf16<true><<<grid, 256, 0, stream>>>(xbf, gwb, gb, ssm, ybf, nullptr);

    // GEMM2: out = y @ out_w^T + out_b  (f32 out)
    gemm_bf16<false><<<grid, 256, 0, stream>>>(ybf, owb, ob, nullptr, nullptr, out);
}
